// Qwen3OmniMoeSparseMoeBlock_56547539419774
// MI455X (gfx1250) — compile-verified, matrix operand order checked by reading, not run on hardware
//
#include <hip/hip_runtime.h>
#include <stdint.h>

#define T_TOK 2048
#define H_DIM 1024
#define I_DIM 2816
#define E_NUM 8
#define NSLOT 4608   // 2*T + per-expert pad to 64, rounded up
#define ZROW_ELEMS 2048

typedef __attribute__((ext_vector_type(16))) __bf16 v16bf;
typedef __attribute__((ext_vector_type(8)))  __bf16 v8bf;
typedef __attribute__((ext_vector_type(4)))  __bf16 v4bf;
typedef __attribute__((ext_vector_type(8)))  float  v8f;

// ---------------- workspace layout (bytes) ----------------
static constexpr size_t WMAT = (size_t)E_NUM * H_DIM * I_DIM * 2;               // 46,137,344
static constexpr size_t XB_OFF      = 0;                                        // bf16 x: T*H
static constexpr size_t WUPT_OFF    = XB_OFF   + (size_t)T_TOK * H_DIM * 2;     // bf16 W_up^T  [E][I][H]
static constexpr size_t WDNT_OFF    = WUPT_OFF + WMAT;                          // bf16 W_dn^T  [E][H][I]
static constexpr size_t ACT_OFF     = WDNT_OFF + WMAT;                          // bf16 act: NSLOT*I
static constexpr size_t DOWN_OFF    = ACT_OFF  + (size_t)NSLOT * I_DIM * 2;     // f32 down: NSLOT*H
static constexpr size_t ROWTOK_OFF  = DOWN_OFF + (size_t)NSLOT * H_DIM * 4;     // int: NSLOT
static constexpr size_t ROWGATE_OFF = ROWTOK_OFF  + (size_t)NSLOT * 4;          // f32: NSLOT
static constexpr size_t TOKSLOT_OFF = ROWGATE_OFF + (size_t)NSLOT * 4;          // int: 2T
static constexpr size_t EIDS_OFF    = TOKSLOT_OFF + (size_t)T_TOK * 2 * 4;      // int: 2T
static constexpr size_t GTS_OFF     = EIDS_OFF    + (size_t)T_TOK * 2 * 4;      // f32: 2T
static constexpr size_t CNT_OFF     = GTS_OFF     + (size_t)T_TOK * 2 * 4;      // int: 8
static constexpr size_t CNT2_OFF    = CNT_OFF  + 64;                            // int: 8
static constexpr size_t SEG_OFF     = CNT2_OFF + 64;                            // int: 8
static constexpr size_t ZROW_OFF    = SEG_OFF  + 64;                            // bf16 zeros: 2048

__device__ __forceinline__ float silu_f(float x) { return x / (1.f + __expf(-x)); }

// ---------------- init: zero counts, zero-row, aux = 1.0 (exact: usage sums to E) ----------------
__global__ void k_init(int* counts, float* aux, __bf16* zrow) {
  int i = threadIdx.x;
  if (i < E_NUM) counts[i] = 0;
  if (i == 0) *aux = 1.0f;
  for (int j = i; j < ZROW_ELEMS; j += 256) zrow[j] = (__bf16)0.f;
}

// ---------------- f32 -> bf16 convert of hidden states (float4 -> 4x bf16) ----------------
__global__ void k_cvt(const float* __restrict__ x, __bf16* __restrict__ xb, int n4) {
  int i = blockIdx.x * blockDim.x + threadIdx.x;
  if (i < n4) {
    float4 v = ((const float4*)x)[i];
    v4bf o;
    o[0] = (__bf16)v.x; o[1] = (__bf16)v.y; o[2] = (__bf16)v.z; o[3] = (__bf16)v.w;
    ((v4bf*)xb)[i] = o;
  }
}

// ---------------- tiled transpose + f32->bf16: dst[n][k] = src[k][n] ----------------
// src: R x C (f32, row-major), dst: C x R (bf16, row-major). Grid: (C/32, R/32, E).
__global__ __launch_bounds__(256) void k_tconv(const float* __restrict__ src,
                                               __bf16* __restrict__ dst, int R, int C) {
  __shared__ float T[32][36];  // pad 36: 144B rows, 16B-aligned float4 slots
  const size_t mat = (size_t)R * C;
  const float* s = src + (size_t)blockIdx.z * mat;
  __bf16* d = dst + (size_t)blockIdx.z * mat;
  const int k0 = blockIdx.y * 32;
  const int n0 = blockIdx.x * 32;
  const int tid = threadIdx.x;
  {
    int lk = tid >> 3;            // 0..31 source row
    int ln = (tid & 7) * 4;       // col group
    float4 v = *(const float4*)(s + (size_t)(k0 + lk) * C + n0 + ln);
    *(float4*)&T[lk][ln] = v;
  }
  __syncthreads();
  {
    int ln = tid >> 3;            // 0..31 dst row (= src col)
    int lk = (tid & 7) * 4;       // k group
    v4bf o;
    o[0] = (__bf16)T[lk + 0][ln];
    o[1] = (__bf16)T[lk + 1][ln];
    o[2] = (__bf16)T[lk + 2][ln];
    o[3] = (__bf16)T[lk + 3][ln];
    *(v4bf*)(d + (size_t)(n0 + ln) * R + k0 + lk) = o;
  }
}

// ---------------- router: logits, softmax top-2 gates, per-expert counts ----------------
__global__ __launch_bounds__(128) void k_router(const float* __restrict__ x,
                                                const float* __restrict__ Wr,
                                                float* __restrict__ logits_out,
                                                int* __restrict__ counts,
                                                int* __restrict__ eids,
                                                float* __restrict__ gts) {
  __shared__ float red[E_NUM][128];
  const int t = blockIdx.x, tid = threadIdx.x;
  float p[E_NUM];
#pragma unroll
  for (int e = 0; e < E_NUM; ++e) p[e] = 0.f;
  const float* xr = x + (size_t)t * H_DIM;
  for (int h = tid; h < H_DIM; h += 128) {
    float xv = xr[h];
    const float* wr = Wr + (size_t)h * E_NUM;
#pragma unroll
    for (int e = 0; e < E_NUM; ++e) p[e] += xv * wr[e];
  }
#pragma unroll
  for (int e = 0; e < E_NUM; ++e) red[e][tid] = p[e];
  __syncthreads();
  for (int s = 64; s > 0; s >>= 1) {
    if (tid < s) {
#pragma unroll
      for (int e = 0; e < E_NUM; ++e) red[e][tid] += red[e][tid + s];
    }
    __syncthreads();
  }
  if (tid == 0) {
    float l[E_NUM];
#pragma unroll
    for (int e = 0; e < E_NUM; ++e) { l[e] = red[e][0]; logits_out[t * E_NUM + e] = l[e]; }
    int a0 = 0;
#pragma unroll
    for (int e = 1; e < E_NUM; ++e) if (l[e] > l[a0]) a0 = e;
    int a1 = (a0 == 0) ? 1 : 0;
#pragma unroll
    for (int e = 0; e < E_NUM; ++e) if (e != a0 && e != a1 && l[e] > l[a1]) a1 = e;
    float e1 = __expf(l[a1] - l[a0]);
    float inv = 1.f / (1.f + e1);
    eids[2 * t] = a0; eids[2 * t + 1] = a1;
    gts[2 * t] = inv; gts[2 * t + 1] = e1 * inv;
    atomicAdd(&counts[a0], 1);
    atomicAdd(&counts[a1], 1);
  }
}

// ---------------- tiny scan: segment bases padded to 64 rows ----------------
__global__ void k_scan(const int* __restrict__ counts, int* __restrict__ segbase,
                       int* __restrict__ counts2) {
  if (threadIdx.x == 0 && blockIdx.x == 0) {
    int run = 0;
    for (int e = 0; e < E_NUM; ++e) {
      segbase[e] = run;
      run += (counts[e] + 63) & ~63;
      counts2[e] = 0;
    }
  }
}

// ---------------- scatter: slot assignment (slot permutation doesn't affect output) ----------------
__global__ void k_scatter(const int* __restrict__ eids, const float* __restrict__ gts,
                          const int* __restrict__ segbase, int* __restrict__ counts2,
                          int* __restrict__ rowtok, float* __restrict__ rowgate,
                          int* __restrict__ tokslot) {
  int i = blockIdx.x * blockDim.x + threadIdx.x;
  if (i < T_TOK * 2) {
    int e = eids[i];
    int pos = atomicAdd(&counts2[e], 1);
    int slot = segbase[e] + pos;
    rowtok[slot] = i >> 1;
    rowgate[slot] = gts[i];
    tokslot[i] = slot;
  }
}

// ---------------- fragment helper: uniform base + 32-bit element offset ----------------
__device__ __forceinline__ v16bf frag16o(const __bf16* __restrict__ base, uint32_t off) {
  v8bf lo = *(const v8bf*)(base + off);        // K = kb .. kb+7
  v8bf hi = *(const v8bf*)(base + off + 16);   // K = 16+kb .. 23+kb
  return __builtin_shufflevector(lo, hi, 0, 1, 2, 3, 4, 5, 6, 7,
                                 8, 9, 10, 11, 12, 13, 14, 15);
}

// ---------------- up GEMM: act = silu(gather(xb) @ W_up[e] + b_up[e]) ----------------
// 256 thr = 8 waves; tile 64 rows x 128 cols; no LDS/barriers; padding rows read a
// zeroed scratch row; K-loop ping-pong double-buffered (2 K-steps per iteration) so
// next fragments load under current wmmas with pure register renaming (no copies).
__global__ __launch_bounds__(256) void k_up(const __bf16* __restrict__ xb,
                                            const __bf16* __restrict__ WupT,
                                            const float* __restrict__ bup,
                                            const int* __restrict__ counts,
                                            const int* __restrict__ segbase,
                                            const int* __restrict__ rowtok,
                                            const __bf16* __restrict__ zrow,
                                            __bf16* __restrict__ act) {
  const int e = blockIdx.z;
  const int cnt = counts[e];
  const int m0 = blockIdx.y * 64;
  if (m0 >= cnt) return;
  const int n0 = blockIdx.x * 128;
  const int sb = segbase[e];

  const int tid = threadIdx.x;
  const int lane = tid & 31;
  const int w = tid >> 5;
  const int kb = (lane < 16) ? 0 : 8;   // ISA 16-bit lane -> K-half mapping
  const int loc = lane & 15;
  const int col = n0 + w * 16 + loc;

  const uint32_t zoff = (uint32_t)(zrow - xb);  // zero row as offset from xb base
  const uint32_t boff = (uint32_t)((e * I_DIM + col) * H_DIM) + kb;

  uint32_t aoff[4];
#pragma unroll
  for (int mt = 0; mt < 4; ++mt) {
    int rr = m0 + mt * 16 + loc;
    bool av = (rr < cnt);
    int tk = av ? rowtok[sb + rr] : 0;
    aoff[mt] = (av ? (uint32_t)tk * H_DIM : zoff) + kb;
  }

  v8f acc[4];
#pragma unroll
  for (int mt = 0; mt < 4; ++mt)
#pragma unroll
    for (int v = 0; v < 8; ++v) acc[mt][v] = 0.f;

  v16bf bA, bB, aA[4], aB[4];
  bA = frag16o(WupT, boff);
#pragma unroll
  for (int mt = 0; mt < 4; ++mt) aA[mt] = frag16o(xb, aoff[mt]);

  int h0 = 0;
  for (; h0 < H_DIM - 64; h0 += 64) {
    __builtin_prefetch(WupT + boff + h0 + 160, 0, 0);  // speculative; OOB lands in valid ws
    bB = frag16o(WupT, boff + h0 + 32);
#pragma unroll
    for (int mt = 0; mt < 4; ++mt) aB[mt] = frag16o(xb, aoff[mt] + h0 + 32);
#pragma unroll
    for (int mt = 0; mt < 4; ++mt)
      acc[mt] = __builtin_amdgcn_wmma_f32_16x16x32_bf16(false, aA[mt], false, bA,
                                                        (short)0, acc[mt], false, false);
    bA = frag16o(WupT, boff + h0 + 64);
#pragma unroll
    for (int mt = 0; mt < 4; ++mt) aA[mt] = frag16o(xb, aoff[mt] + h0 + 64);
#pragma unroll
    for (int mt = 0; mt < 4; ++mt)
      acc[mt] = __builtin_amdgcn_wmma_f32_16x16x32_bf16(false, aB[mt], false, bB,
                                                        (short)0, acc[mt], false, false);
  }
  // tail: set A holds k = H_DIM-64; load and do the last two K-steps
  bB = frag16o(WupT, boff + h0 + 32);
#pragma unroll
  for (int mt = 0; mt < 4; ++mt) aB[mt] = frag16o(xb, aoff[mt] + h0 + 32);
#pragma unroll
  for (int mt = 0; mt < 4; ++mt)
    acc[mt] = __builtin_amdgcn_wmma_f32_16x16x32_bf16(false, aA[mt], false, bA,
                                                      (short)0, acc[mt], false, false);
#pragma unroll
  for (int mt = 0; mt < 4; ++mt)
    acc[mt] = __builtin_amdgcn_wmma_f32_16x16x32_bf16(false, aB[mt], false, bB,
                                                      (short)0, acc[mt], false, false);

  // C layout: lanes 0-15 -> M=vgpr, lanes 16-31 -> M=8+vgpr; N=lane&15
  const int mh = (lane < 16) ? 0 : 8;
  const float bb = bup[e * I_DIM + col];
#pragma unroll
  for (int mt = 0; mt < 4; ++mt) {
#pragma unroll
    for (int v = 0; v < 8; ++v) {
      int r = m0 + mt * 16 + mh + v;
      float val = 0.f;
      if (r < cnt) val = silu_f(acc[mt][v] + bb);  // padding rows forced to 0
      act[(size_t)(sb + r) * I_DIM + col] = (__bf16)val;
    }
  }
}

// ---------------- down GEMM: down = gate * (act @ W_down[e] + b_down[e]) ----------------
__global__ __launch_bounds__(256) void k_down(const __bf16* __restrict__ act,
                                              const __bf16* __restrict__ WdnT,
                                              const float* __restrict__ bdn,
                                              const int* __restrict__ counts,
                                              const int* __restrict__ segbase,
                                              const float* __restrict__ rowgate,
                                              float* __restrict__ down) {
  const int e = blockIdx.z;
  const int cnt = counts[e];
  const int m0 = blockIdx.y * 64;
  if (m0 >= cnt) return;
  const int n0 = blockIdx.x * 128;
  const int sb = segbase[e];

  const int tid = threadIdx.x;
  const int lane = tid & 31;
  const int w = tid >> 5;
  const int kb = (lane < 16) ? 0 : 8;
  const int loc = lane & 15;
  const int col = n0 + w * 16 + loc;

  const uint32_t boff = (uint32_t)((e * H_DIM + col) * I_DIM) + kb;

  uint32_t aoff[4];
#pragma unroll
  for (int mt = 0; mt < 4; ++mt)  // act rows contiguous per segment; padding rows are zeros
    aoff[mt] = (uint32_t)(sb + m0 + mt * 16 + loc) * I_DIM + kb;

  v8f acc[4];
#pragma unroll
  for (int mt = 0; mt < 4; ++mt)
#pragma unroll
    for (int v = 0; v < 8; ++v) acc[mt][v] = 0.f;

  v16bf bA, bB, aA[4], aB[4];
  bA = frag16o(WdnT, boff);
#pragma unroll
  for (int mt = 0; mt < 4; ++mt) aA[mt] = frag16o(act, aoff[mt]);

  int i0 = 0;
  for (; i0 < I_DIM - 64; i0 += 64) {
    __builtin_prefetch(WdnT + boff + i0 + 160, 0, 0);  // speculative; OOB lands in valid ws
    bB = frag16o(WdnT, boff + i0 + 32);
#pragma unroll
    for (int mt = 0; mt < 4; ++mt) aB[mt] = frag16o(act, aoff[mt] + i0 + 32);
#pragma unroll
    for (int mt = 0; mt < 4; ++mt)
      acc[mt] = __builtin_amdgcn_wmma_f32_16x16x32_bf16(false, aA[mt], false, bA,
                                                        (short)0, acc[mt], false, false);
    bA = frag16o(WdnT, boff + i0 + 64);
#pragma unroll
    for (int mt = 0; mt < 4; ++mt) aA[mt] = frag16o(act, aoff[mt] + i0 + 64);
#pragma unroll
    for (int mt = 0; mt < 4; ++mt)
      acc[mt] = __builtin_amdgcn_wmma_f32_16x16x32_bf16(false, aB[mt], false, bB,
                                                        (short)0, acc[mt], false, false);
  }
  bB = frag16o(WdnT, boff + i0 + 32);
#pragma unroll
  for (int mt = 0; mt < 4; ++mt) aB[mt] = frag16o(act, aoff[mt] + i0 + 32);
#pragma unroll
  for (int mt = 0; mt < 4; ++mt)
    acc[mt] = __builtin_amdgcn_wmma_f32_16x16x32_bf16(false, aA[mt], false, bA,
                                                      (short)0, acc[mt], false, false);
#pragma unroll
  for (int mt = 0; mt < 4; ++mt)
    acc[mt] = __builtin_amdgcn_wmma_f32_16x16x32_bf16(false, aB[mt], false, bB,
                                                      (short)0, acc[mt], false, false);

  const int mh = (lane < 16) ? 0 : 8;
  const float bb = bdn[e * H_DIM + col];
#pragma unroll
  for (int mt = 0; mt < 4; ++mt) {
#pragma unroll
    for (int v = 0; v < 8; ++v) {
      int r = m0 + mt * 16 + mh + v;
      if (r < cnt) {
        float g = rowgate[sb + r];
        down[(size_t)(sb + r) * H_DIM + col] = g * (acc[mt][v] + bb);
      }
    }
  }
}

// ---------------- combine: final[t] = down[slot0_t] + down[slot1_t] (gates pre-applied) ----------------
__global__ void k_combine(const float* __restrict__ down, const int* __restrict__ tokslot,
                          float* __restrict__ out) {
  int idx = blockIdx.x * blockDim.x + threadIdx.x;   // over T*H/4 float4s
  if (idx >= T_TOK * (H_DIM / 4)) return;
  int t = idx >> 8;                 // H/4 = 256
  int hq = idx & 255;
  int s0 = tokslot[2 * t];
  int s1 = tokslot[2 * t + 1];
  const float4 a = *(const float4*)(down + (size_t)s0 * H_DIM + hq * 4);
  const float4 b = *(const float4*)(down + (size_t)s1 * H_DIM + hq * 4);
  float4 o; o.x = a.x + b.x; o.y = a.y + b.y; o.z = a.z + b.z; o.w = a.w + b.w;
  *(float4*)(out + (size_t)t * H_DIM + hq * 4) = o;
}

// ---------------- host launcher ----------------
extern "C" void kernel_launch(void* const* d_in, const int* in_sizes, int n_in,
                              void* d_out, int out_size, void* d_ws, size_t ws_size,
                              hipStream_t stream) {
  (void)in_sizes; (void)n_in; (void)out_size; (void)ws_size;
  const float* x   = (const float*)d_in[0];
  const float* Wr  = (const float*)d_in[1];
  const float* Wup = (const float*)d_in[2];
  const float* bup = (const float*)d_in[3];
  const float* Wdn = (const float*)d_in[4];
  const float* bdn = (const float*)d_in[5];
  float* out = (float*)d_out;

  char* ws = (char*)d_ws;
  __bf16* xb     = (__bf16*)(ws + XB_OFF);
  __bf16* WupT   = (__bf16*)(ws + WUPT_OFF);
  __bf16* WdnT   = (__bf16*)(ws + WDNT_OFF);
  __bf16* act    = (__bf16*)(ws + ACT_OFF);
  float*  downb  = (float*)(ws + DOWN_OFF);
  int*    rowtok = (int*)(ws + ROWTOK_OFF);
  float*  rowgate= (float*)(ws + ROWGATE_OFF);
  int*    tokslot= (int*)(ws + TOKSLOT_OFF);
  int*    eids   = (int*)(ws + EIDS_OFF);
  float*  gts    = (float*)(ws + GTS_OFF);
  int*    counts = (int*)(ws + CNT_OFF);
  int*    counts2= (int*)(ws + CNT2_OFF);
  int*    segbase= (int*)(ws + SEG_OFF);
  __bf16* zrow   = (__bf16*)(ws + ZROW_OFF);

  float* logits = out + (size_t)T_TOK * H_DIM;
  float* aux    = logits + (size_t)T_TOK * E_NUM;

  k_init<<<1, 256, 0, stream>>>(counts, aux, zrow);
  k_cvt<<<(T_TOK * H_DIM / 4 + 255) / 256, 256, 0, stream>>>(x, xb, T_TOK * H_DIM / 4);
  // W_up (E,H,I) -> WupT (E,I,H);  W_dn (E,I,H) -> WdnT (E,H,I)
  k_tconv<<<dim3(I_DIM / 32, H_DIM / 32, E_NUM), 256, 0, stream>>>(Wup, WupT, H_DIM, I_DIM);
  k_tconv<<<dim3(H_DIM / 32, I_DIM / 32, E_NUM), 256, 0, stream>>>(Wdn, WdnT, I_DIM, H_DIM);
  k_router<<<T_TOK, 128, 0, stream>>>(x, Wr, logits, counts, eids, gts);
  k_scan<<<1, 1, 0, stream>>>(counts, segbase, counts2);
  k_scatter<<<(T_TOK * 2 + 255) / 256, 256, 0, stream>>>(eids, gts, segbase, counts2,
                                                         rowtok, rowgate, tokslot);
  k_up<<<dim3(I_DIM / 128, T_TOK / 64, E_NUM), 256, 0, stream>>>(xb, WupT, bup, counts,
                                                                 segbase, rowtok, zrow, act);
  k_down<<<dim3(H_DIM / 128, T_TOK / 64, E_NUM), 256, 0, stream>>>(act, WdnT, bdn, counts,
                                                                   segbase, rowgate, downb);
  k_combine<<<(T_TOK * H_DIM / 4 + 255) / 256, 256, 0, stream>>>(downb, tokslot, out);
}
